// Atten_Spa_8100308321064
// MI455X (gfx1250) — compile-verified, matrix operand order checked by reading, not run on hardware
//
#include <hip/hip_runtime.h>
#include <cstdint>

// Spatial self-attention (SAGAN) for B=4, C=256, H=W=64 (N=4096), d=C/8=32.
// Flash-style two-pass softmax; all GEMMs on v_wmma_f32_16x16x32_f16.

typedef __attribute__((ext_vector_type(16))) _Float16 v16h;
typedef __attribute__((ext_vector_type(8)))  float    v8f;

#define B_   4
#define C_   256
#define C8_  32
#define HN_  4096       // N = H*W
#define LDSP 36         // padded LDS row stride (f16): 72B -> 8B aligned, conflict-free

union FragH {
  v16h     h;
  uint32_t u[8];
  uint2    u2[4];
  uint4    q[2];
  _Float16 e[16];
};

__device__ __forceinline__ v8f wmma_f16(const FragH& a, const FragH& b, v8f c) {
  return __builtin_amdgcn_wmma_f32_16x16x32_f16(false, a.h, false, b.h,
                                                (short)0, c, false, false);
}

// Row-major f16 fragment load (A-pattern; also B-pattern applied to B^T rows).
// Reads f16 offsets [koff..koff+7] and [16+koff..16+koff+7]; both 16B aligned.
__device__ __forceinline__ void load_frag(FragH& f, const _Float16* __restrict__ row,
                                          int koff) {
  f.q[0] = *(const uint4*)(row + koff);
  f.q[1] = *(const uint4*)(row + 16 + koff);
}

// ---------------------------------------------------------------------------
// Projection: P = W @ X + bias.  W:[Dp][C_] f32, X:[C_][N] f32 per batch.
// transposed=1 -> store P^T as [B][N][Dp] f16 (Q, K);  else [B][Dp][N] (V).
// One wave per 16x16 output tile, K-loop over C_ in steps of 32.
// ---------------------------------------------------------------------------
__global__ void __launch_bounds__(256)
proj_kernel(const float* __restrict__ X,
            const float* __restrict__ W,
            const float* __restrict__ bias,
            _Float16* __restrict__ Out,
            int Dp, int transposed) {
  const int wave = blockIdx.x * (blockDim.x >> 5) + (threadIdx.x >> 5);
  const int lane = threadIdx.x & 31;
  const int l15  = lane & 15;
  const int hi   = lane >> 4;
  const int koff = hi * 8;

  const int ntiles = HN_ / 16;
  const int dtiles = Dp / 16;
  const int tilesPerBatch = ntiles * dtiles;
  const int b  = wave / tilesPerBatch;
  const int t  = wave % tilesPerBatch;
  const int d0 = (t / ntiles) * 16;
  const int n0 = (t % ntiles) * 16;

  const float* Xb = X + (size_t)b * C_ * HN_;

  v8f acc = {};
  for (int c0 = 0; c0 < C_; c0 += 32) {
    FragH a, bf;
    // A fragment: W tile, rows d (=lane&15), contiguous K pairs along C
    const float* wr = W + (size_t)(d0 + l15) * C_ + c0;
#pragma unroll
    for (int j = 0; j < 4; ++j) {
      a.e[2*j]     = (_Float16)wr[2*j + koff];
      a.e[2*j+1]   = (_Float16)wr[2*j + koff + 1];
      a.e[8+2*j]   = (_Float16)wr[16 + 2*j + koff];
      a.e[8+2*j+1] = (_Float16)wr[16 + 2*j + koff + 1];
    }
    // B fragment: reads B^T rows (n = lane&15), K strided by HN_
    const float* xc = Xb + n0 + l15;
#pragma unroll
    for (int j = 0; j < 4; ++j) {
      bf.e[2*j]     = (_Float16)xc[(size_t)(c0 + 2*j + koff) * HN_];
      bf.e[2*j+1]   = (_Float16)xc[(size_t)(c0 + 2*j + koff + 1) * HN_];
      bf.e[8+2*j]   = (_Float16)xc[(size_t)(c0 + 16 + 2*j + koff) * HN_];
      bf.e[8+2*j+1] = (_Float16)xc[(size_t)(c0 + 16 + 2*j + koff + 1) * HN_];
    }
    acc = wmma_f16(a, bf, acc);
  }

  // D layout: lane holds (d = d0 + r + 8*hi, n = n0 + (lane&15))
#pragma unroll
  for (int r = 0; r < 8; ++r) {
    const int d = d0 + r + hi * 8;
    const float v = acc[r] + bias[d];
    if (transposed)
      Out[((size_t)b * HN_ + (n0 + l15)) * Dp + d] = (_Float16)v;
    else
      Out[((size_t)b * Dp + d) * HN_ + (n0 + l15)] = (_Float16)v;
  }
}

// ---------------------------------------------------------------------------
// Pass 1: per query row, online row-max M and row-sum L of exp(E - M).
// One wave per 16-query-row tile; streams K tiles of 16.
// ---------------------------------------------------------------------------
__global__ void __launch_bounds__(256)
attn_stats_kernel(const _Float16* __restrict__ Qh,   // [B][N][32]
                  const _Float16* __restrict__ Kh,   // [B][N][32]
                  float* __restrict__ Mrow,
                  float* __restrict__ Lrow) {
  const int wave = blockIdx.x * (blockDim.x >> 5) + (threadIdx.x >> 5);
  const int lane = threadIdx.x & 31;
  const int l15 = lane & 15, hi = lane >> 4, koff = hi * 8;
  const int b  = wave >> 8;          // / 256 tiles per batch
  const int n0 = (wave & 255) << 4;

  FragH qa;
  load_frag(qa, Qh + ((size_t)b * HN_ + n0 + l15) * C8_, koff);

  float M[8], S[8];
#pragma unroll
  for (int r = 0; r < 8; ++r) { M[r] = -1e30f; S[r] = 0.f; }

  const _Float16* Kb = Kh + (size_t)b * HN_ * C8_;
  for (int m0 = 0; m0 < HN_; m0 += 16) {
    FragH kb;
    load_frag(kb, Kb + (size_t)(m0 + l15) * C8_, koff);
    v8f e = {};
    e = wmma_f16(qa, kb, e);
#pragma unroll
    for (int r = 0; r < 8; ++r) {
      float t = e[r];
      float rm = t;                                   // row-max over 16 lanes
      rm = fmaxf(rm, __shfl_xor(rm, 1, 16));
      rm = fmaxf(rm, __shfl_xor(rm, 2, 16));
      rm = fmaxf(rm, __shfl_xor(rm, 4, 16));
      rm = fmaxf(rm, __shfl_xor(rm, 8, 16));
      const float nm = fmaxf(M[r], rm);
      float ex = __expf(t - nm);                      // row-sum over 16 lanes
      ex += __shfl_xor(ex, 1, 16);
      ex += __shfl_xor(ex, 2, 16);
      ex += __shfl_xor(ex, 4, 16);
      ex += __shfl_xor(ex, 8, 16);
      S[r] = S[r] * __expf(M[r] - nm) + ex;
      M[r] = nm;
    }
  }

  if (l15 == 0) {
#pragma unroll
    for (int r = 0; r < 8; ++r) {
      const int n = n0 + r + hi * 8;
      Mrow[(size_t)b * HN_ + n] = M[r];
      Lrow[(size_t)b * HN_ + n] = S[r];
    }
  }
}

// ---------------------------------------------------------------------------
// Pass 2: out[c][n] = gamma * (sum_m exp(E-M) * V[c][m]) / L + x[c][n].
// One wave per (16 query rows x 64 channels) tile; m in steps of 32.
// E in D-layout -> exp -> LDS tile -> reload as B-fragment for PV WMMA.
// ---------------------------------------------------------------------------
__global__ void __launch_bounds__(256)
attn_out_kernel(const _Float16* __restrict__ Qh,
                const _Float16* __restrict__ Kh,
                const _Float16* __restrict__ Vh,   // [B][C][N]
                const float* __restrict__ Mrow,
                const float* __restrict__ Lrow,
                const float* __restrict__ X,
                const float* __restrict__ gamma,
                float* __restrict__ Out) {
  __shared__ _Float16 lds[8][16 * LDSP];
  const int wv   = threadIdx.x >> 5;
  const int wave = blockIdx.x * (blockDim.x >> 5) + wv;
  const int lane = threadIdx.x & 31;
  const int l15 = lane & 15, hi = lane >> 4, koff = hi * 8;

  const int b     = wave >> 10;              // / (256 tiles * 4 chunks)
  const int rem   = wave & 1023;
  const int n0    = (rem >> 2) << 4;
  const int cbase = (rem & 3) << 6;          // channel chunk of 64

  FragH qa;
  load_frag(qa, Qh + ((size_t)b * HN_ + n0 + l15) * C8_, koff);

  float mst[8];
#pragma unroll
  for (int r = 0; r < 8; ++r)
    mst[r] = Mrow[(size_t)b * HN_ + n0 + r + hi * 8];

  v8f acc[4] = {};
  const _Float16* Kb = Kh + (size_t)b * HN_ * C8_;
  const _Float16* Vb = Vh + (size_t)b * C_ * HN_;
  _Float16* myl = lds[wv];

  for (int m0 = 0; m0 < HN_; m0 += 32) {
    // Two 16-wide energy tiles -> exp -> LDS P tile [16 n][32 m]
#pragma unroll
    for (int half = 0; half < 2; ++half) {
      FragH kb;
      load_frag(kb, Kb + (size_t)(m0 + half * 16 + l15) * C8_, koff);
      v8f e = {};
      e = wmma_f16(qa, kb, e);
#pragma unroll
      for (int r = 0; r < 8; ++r) {
        const float p = __expf(e[r] - mst[r]);
        myl[(r + hi * 8) * LDSP + half * 16 + l15] = (_Float16)p;
      }
    }
    asm volatile("s_wait_dscnt 0" ::: "memory");   // LDS RAW fence (same wave)

    // P as B-fragment: rows n = lane&15, K pairs along m (4x ds_load_b64)
    FragH pb;
    {
      const _Float16* pr = myl + l15 * LDSP;
      pb.u2[0] = *(const uint2*)(pr + koff);
      pb.u2[1] = *(const uint2*)(pr + koff + 4);
      pb.u2[2] = *(const uint2*)(pr + 16 + koff);
      pb.u2[3] = *(const uint2*)(pr + 16 + koff + 4);
    }
    // V as A-fragment: rows c = lane&15 within c-tile, contiguous m pairs
#pragma unroll
    for (int t = 0; t < 4; ++t) {
      FragH va;
      load_frag(va, Vb + (size_t)(cbase + t * 16 + l15) * HN_ + m0, koff);
      acc[t] = wmma_f16(va, pb, acc[t]);
    }
    asm volatile("" ::: "memory");  // keep next-iter LDS stores after P loads
  }

  // Epilogue: scale by 1/L (per n-column == per lane) and add residual.
  const float ln = 1.0f / Lrow[(size_t)b * HN_ + n0 + l15];
  const float g  = gamma[0];
#pragma unroll
  for (int t = 0; t < 4; ++t) {
#pragma unroll
    for (int r = 0; r < 8; ++r) {
      const int c = cbase + t * 16 + r + hi * 8;
      const size_t idx = ((size_t)b * C_ + c) * HN_ + n0 + l15;
      Out[idx] = g * (acc[t][r] * ln) + X[idx];
    }
  }
}

// ---------------------------------------------------------------------------
extern "C" void kernel_launch(void* const* d_in, const int* in_sizes, int n_in,
                              void* d_out, int out_size, void* d_ws, size_t ws_size,
                              hipStream_t stream) {
  (void)in_sizes; (void)n_in; (void)out_size; (void)ws_size;
  const float* x     = (const float*)d_in[0];
  const float* wq    = (const float*)d_in[1];
  const float* bq    = (const float*)d_in[2];
  const float* wk    = (const float*)d_in[3];
  const float* bk    = (const float*)d_in[4];
  const float* wv    = (const float*)d_in[5];
  const float* bv    = (const float*)d_in[6];
  const float* gamma = (const float*)d_in[7];
  float* out = (float*)d_out;

  // Workspace layout (bytes):
  //   Qh f16 [B][N][32] : 1 MiB
  //   Kh f16 [B][N][32] : 1 MiB
  //   Vh f16 [B][C][N]  : 8 MiB
  //   Mrow f32 [B][N]   : 64 KiB
  //   Lrow f32 [B][N]   : 64 KiB
  char* ws = (char*)d_ws;
  _Float16* Qh = (_Float16*)(ws);
  _Float16* Kh = (_Float16*)(ws + (1u << 20));
  _Float16* Vh = (_Float16*)(ws + (2u << 20));
  float* Mrow  = (float*)(ws + 10485760u);
  float* Lrow  = (float*)(ws + 10485760u + 65536u);

  const dim3 blk(256);  // 8 waves/block (wave32)

  // Projections: waves = B * (Dp/16) * (N/16); 8 waves per block.
  proj_kernel<<<256,  blk, 0, stream>>>(x, wq, bq, Qh, C8_, 1);
  proj_kernel<<<256,  blk, 0, stream>>>(x, wk, bk, Kh, C8_, 1);
  proj_kernel<<<2048, blk, 0, stream>>>(x, wv, bv, Vh, C_,  0);

  // Softmax stats: 4 * 256 tiles = 1024 waves.
  attn_stats_kernel<<<128, blk, 0, stream>>>(Qh, Kh, Mrow, Lrow);

  // PV pass: 4 * 256 tiles * 4 channel-chunks = 4096 waves.
  attn_out_kernel<<<512, blk, 0, stream>>>(Qh, Kh, Vh, Mrow, Lrow, x, gamma, out);
}
